// RNN_2259152798289
// MI455X (gfx1250) — matrix operand order repaired and checked by reading, not compiled, not run on hardware
//
#include <hip/hip_runtime.h>

// LSTM: B=32, T=512, D=512, H=1024.
// Per step: z = [x_t | h] @ [Wx;Wh] + b via v_wmma_f32_16x16x32_bf16.
// h state is staged to LDS with global_load_async_to_lds_b128 (ASYNCcnt),
// overlapped with the X-phase WMMAs; weights (12 MB bf16) stay L2-resident.

#define B_   32
#define T_   512
#define D_   512
#define H_   1024
#define KC_  (D_ + H_)   // 1536 combined reduction dim
#define N4H_ (4 * H_)    // 4096 gate columns (i,f,g,o)

#define HROW_ 1032       // padded LDS row stride (elems): +16B to spread banks

typedef __attribute__((ext_vector_type(16))) __bf16 v16bf;
typedef __attribute__((ext_vector_type(8)))  float  v8f;

union BF16x16 {
    v16bf v;
    uint4 q[2];
};

__device__ __forceinline__ unsigned short f32_to_bf16(float f) {
    union { float f; unsigned int u; } x;
    x.f = f;
    unsigned int u = x.u;
    unsigned int r = u + 0x7FFFu + ((u >> 16) & 1u);   // round-to-nearest-even
    if ((u & 0x7F800000u) == 0x7F800000u) r = u;       // Inf/NaN passthrough
    return (unsigned short)(r >> 16);
}

// ---- prep: fp32 -> bf16 copy of the inputs ----
__global__ void cvt_x_bf16(const float* __restrict__ x,
                           unsigned short* __restrict__ xb, int n) {
    int i = blockIdx.x * blockDim.x + threadIdx.x;
    if (i < n) xb[i] = f32_to_bf16(x[i]);
}

// ---- prep: WT[n][k] = bf16( k<D ? Wx[k][n] : Wh[k-D][n] )  (transposed) ----
__global__ void build_wt(const float* __restrict__ Wx,
                         const float* __restrict__ Wh,
                         unsigned short* __restrict__ wt) {
    int i = blockIdx.x * blockDim.x + threadIdx.x;
    if (i >= N4H_ * KC_) return;
    int n = i / KC_;
    int k = i - n * KC_;
    float v = (k < D_) ? Wx[(size_t)k * N4H_ + n]
                       : Wh[(size_t)(k - D_) * N4H_ + n];
    wt[i] = f32_to_bf16(v);
}

// ---- prep: zero h (bf16) and c (fp32) state ----
__global__ void zero_state(unsigned short* __restrict__ hb,
                           float* __restrict__ c, int n) {
    int i = blockIdx.x * blockDim.x + threadIdx.x;
    if (i < n) { hb[i] = 0; c[i] = 0.0f; }
}

// ---- one recurrence step ----
// grid = H/16 blocks, 256 threads = 8 waves: wave = (gate<<1)|mtile.
__global__ __launch_bounds__(256)
void lstm_step(const unsigned short* __restrict__ xb,   // [B,T,D] bf16
               const unsigned short* __restrict__ wt,   // [4096][1536] bf16
               const float* __restrict__ bias,          // [4096] fp32
               const int*   __restrict__ seqlen,        // [B]
               unsigned short* __restrict__ hb,         // [B,H] bf16 state
               float* __restrict__ c,                   // [B,H] fp32 state
               float* __restrict__ out,                 // [B,H] fp32 result
               int t) {
    __shared__ unsigned short hsh[B_ * HROW_];          // 64.5 KB h tile
    __shared__ float zs[4][B_][16];                     // 8 KB z tiles

    const int tid   = threadIdx.x;
    const int wave  = tid >> 5;
    const int lane  = tid & 31;
    const int mt    = wave & 1;        // batch rows 0-15 / 16-31
    const int gate  = wave >> 1;       // 0..3 = i,f,g,o
    const int col0  = blockIdx.x * 16; // h-column tile

    const int lrow  = lane & 15;
    const int lhalf = lane >> 4;
    const int arow  = mt * 16 + lrow;             // A fragment: batch row
    const int akoff = lhalf * 8;                  // ISA 16-bit A layout
    const int ncol  = gate * H_ + col0 + lrow;    // B fragment: gate column
    const int bkoff = lhalf * 8;                  // ISA 16-bit B layout

    // ---- kick off async copy of the whole h state (32 x 1024 bf16) to LDS;
    //      ASYNCcnt tracks it while the X-phase WMMAs run. 4096 x 16B moves.
    {
        const unsigned lbase = (unsigned)(size_t)(&hsh[0]);
        const unsigned long long gbase = (unsigned long long)(size_t)hb;
        #pragma unroll
        for (int i = 0; i < 16; ++i) {
            int f   = tid + i * 256;              // 0..4095
            int row = f >> 7;                     // batch row
            int seg = f & 127;                    // 16B segment in row
            unsigned lds = lbase + (unsigned)(row * (HROW_ * 2) + seg * 16);
            unsigned long long g = gbase + (unsigned long long)(row * 2048 + seg * 16);
            asm volatile("global_load_async_to_lds_b128 %0, %1, off"
                         :: "v"(lds), "v"(g) : "memory");
        }
    }

    v8f acc = {};

    // ---- X phase: K = 0..511 from bf16 inputs (overlaps the async copy) ----
    {
        const unsigned short* xp = xb + ((size_t)arow * T_ + t) * D_ + akoff;
        const unsigned short* bp = wt + (size_t)ncol * KC_ + bkoff;
        #pragma unroll 4
        for (int k0 = 0; k0 < D_; k0 += 32) {
            BF16x16 a, b;
            a.q[0] = *(const uint4*)(xp + k0);
            a.q[1] = *(const uint4*)(xp + k0 + 16);
            b.q[0] = *(const uint4*)(bp + k0);
            b.q[1] = *(const uint4*)(bp + k0 + 16);
            __builtin_prefetch(bp + k0 + 32, 0, 1);     // global_prefetch_b8
            acc = __builtin_amdgcn_wmma_f32_16x16x32_bf16(
                      false, a.v, false, b.v, (short)0, acc, false, false);
        }
    }

    // ---- wait for the async h copy, make it visible block-wide ----
    asm volatile("s_wait_asynccnt 0x0" ::: "memory");
    __syncthreads();

    // ---- H phase: K = 512..1535, A fragments from LDS (ds_load_b128) ----
    {
        const unsigned short* hp = &hsh[(unsigned)arow * HROW_ + akoff];
        const unsigned short* bp = wt + (size_t)ncol * KC_ + D_ + bkoff;
        #pragma unroll 4
        for (int k0 = 0; k0 < H_; k0 += 32) {
            BF16x16 a, b;
            a.q[0] = *(const uint4*)(hp + k0);
            a.q[1] = *(const uint4*)(hp + k0 + 16);
            b.q[0] = *(const uint4*)(bp + k0);
            b.q[1] = *(const uint4*)(bp + k0 + 16);
            if (k0 + 32 < H_) __builtin_prefetch(bp + k0 + 32, 0, 1);
            acc = __builtin_amdgcn_wmma_f32_16x16x32_bf16(
                      false, a.v, false, b.v, (short)0, acc, false, false);
        }
    }

    // ---- epilogue: bias + stash 16x16 fp32 tile per (gate, batch-half) ----
    // C layout: VGPR e -> M = e + 8*lhalf, N = lrow.
    const float bv = bias[ncol];
    #pragma unroll
    for (int e = 0; e < 8; ++e) {
        int m = mt * 16 + lhalf * 8 + e;
        zs[gate][m][lrow] = acc[e] + bv;
    }
    __syncthreads();

    // ---- fused gate nonlinearities + state update (512 elems / 256 thr) ----
    for (int e = tid; e < B_ * 16; e += 256) {
        int m = e >> 4;
        int n = e & 15;
        int hcol = col0 + n;
        float zi = zs[0][m][n];
        float zf = zs[1][m][n];
        float zg = zs[2][m][n];
        float zo = zs[3][m][n];
        float ig = 1.0f / (1.0f + __expf(-zi));
        float fg = 1.0f / (1.0f + __expf(-zf));
        float og = 1.0f / (1.0f + __expf(-zo));
        float gg = tanhf(zg);
        size_t idx = (size_t)m * H_ + hcol;
        float cn = fg * c[idx] + ig * gg;
        float hn = og * tanhf(cn);
        c[idx]  = cn;
        hb[idx] = f32_to_bf16(hn);
        if (seqlen[m] - 1 == t) out[idx] = hn;   // gather at last valid step
    }
}

extern "C" void kernel_launch(void* const* d_in, const int* in_sizes, int n_in,
                              void* d_out, int out_size, void* d_ws, size_t ws_size,
                              hipStream_t stream) {
    const float* x    = (const float*)d_in[0];
    const int*   sl   = (const int*)  d_in[1];
    const float* Wx   = (const float*)d_in[2];
    const float* Wh   = (const float*)d_in[3];
    const float* bias = (const float*)d_in[4];
    float* out = (float*)d_out;

    // Workspace carve-out (all offsets 256B-aligned), total ~29.6 MB.
    char* ws = (char*)d_ws;
    unsigned short* xb = (unsigned short*)ws;                       // 16 MB
    unsigned short* wt = (unsigned short*)(ws + (size_t)B_ * T_ * D_ * 2);
    unsigned short* hb = (unsigned short*)((char*)wt + (size_t)N4H_ * KC_ * 2);
    float*          c  = (float*)((char*)hb + (size_t)B_ * H_ * 2);

    const int nx = B_ * T_ * D_;
    cvt_x_bf16<<<(nx + 255) / 256, 256, 0, stream>>>(x, xb, nx);

    const int nw = N4H_ * KC_;
    build_wt<<<(nw + 255) / 256, 256, 0, stream>>>(Wx, Wh, wt);

    zero_state<<<(B_ * H_ + 255) / 256, 256, 0, stream>>>(hb, c, B_ * H_);

    // 512 sequential steps; 12 MB bf16 weights stay L2-resident across steps.
    for (int t = 0; t < T_; ++t) {
        lstm_step<<<H_ / 16, 256, 0, stream>>>(xb, wt, bias, sl, hb, c, out, t);
    }
}
